// FDKConeBeamReconstruction_80281528696890
// MI455X (gfx1250) — compile-verified, matrix operand order checked by reading, not run on hardware
//
#include <hip/hip_runtime.h>
#include <hip/hip_bf16.h>

// ---------------------------------------------------------------------------
// FDK cone-beam reconstruction for MI455X (gfx1250, wave32, WMMA).
//
// Ramp filtering == circular convolution == GEMM vs a circulant matrix,
// executed with v_wmma_f32_16x16x32_f16.  LDS tiles are stored in
// *fragment order* so each lane loads its WMMA operand as one contiguous
// 32-byte chunk (2x ds_load_b128) instead of 16 strided u16 reads.
// Pre-weighting (cosine + redundancy) is fused into A-tile staging.
// Backprojection is a coalesced gather kernel.
// ---------------------------------------------------------------------------

#define DET_U   384
#define DET_V   256
#define NANG    256
#define NFREQ   193        // rfft bins of 384
#define VOLN    128
#define D_SO_F  500.0f
#define D_SD_F  1000.0f
#define PI_F    3.14159265358979323846f

typedef __attribute__((ext_vector_type(16))) _Float16 v16h;
typedef __attribute__((ext_vector_type(8)))  _Float16 v8h;
typedef __attribute__((ext_vector_type(8)))  float    v8f;

// Workspace offsets (bytes)
#define OFF_H    0u            // h[384] f32
#define OFF_HM   4096u         // Hf16[384*384]
#define OFF_FILT (512u*1024u)  // filtered f32 [65536*384] = 100663296 B

// Fragment-order position helpers (16-bit 16x32 WMMA operand layout).
//   lane = (m or n) + 16*((k>>3)&1) ;  idx = ((k>>4)<<3) | (k&7)
__device__ __forceinline__ int frag_lane(int mn, int k) {
    return mn + 16 * ((k >> 3) & 1);
}
__device__ __forceinline__ int frag_idx(int k) {
    return ((k >> 4) << 3) | (k & 7);
}

// ---------------------------------------------------------------------------
// Kernel 1: h[n] = irfft(ramp)[n]  (exact integer phase reduction mod 384)
// ---------------------------------------------------------------------------
__global__ void build_h(const float* __restrict__ ramp, float* __restrict__ h) {
    const int n = threadIdx.x;              // 0..383, one block of 384
    const float w0 = 2.0f * PI_F / (float)DET_U;
    float sum = ramp[0];
    for (int k = 1; k < NFREQ - 1; ++k) {
        int kn = (k * n) % DET_U;
        sum += 2.0f * ramp[k] * cosf(w0 * (float)kn);
    }
    sum += ramp[NFREQ - 1] * ((n & 1) ? -1.0f : 1.0f);   // Nyquist: cos(pi*n)
    h[n] = sum / (float)DET_U;
}

// ---------------------------------------------------------------------------
// Kernel 2: circulant matrix H[k][u] = (f16) h[(u - k) mod 384]
// ---------------------------------------------------------------------------
__global__ void build_Hmat(const float* __restrict__ h, _Float16* __restrict__ H) {
    const int k = blockIdx.x;               // 0..383
    const int u = threadIdx.x;              // 0..383
    int d = u - k; if (d < 0) d += DET_U;
    H[(size_t)k * DET_U + u] = (_Float16)h[d];
}

// ---------------------------------------------------------------------------
// Kernel 3: filtered = (proj * cos_w * redw) @ H   via WMMA f16 -> f32
//   M = 65536, K = N = 384.  Block tile 64(M) x 64(N), K-step 32.
//   8 waves; each wave computes a 16x32 strip (2 WMMA tiles, shared A frag).
// ---------------------------------------------------------------------------
#define TILE_M 64
#define TILE_N 64
#define TILE_K 32

__global__ __launch_bounds__(256)
void filter_gemm(const float* __restrict__ proj,
                 const float* __restrict__ redw,
                 const _Float16* __restrict__ H,
                 float* __restrict__ out)
{
    // Fragment-order tiles: [waveTile][lane][16 halfs]
    __shared__ __attribute__((aligned(32))) _Float16 sA[4 * 32 * 16];  // 4 KB
    __shared__ __attribute__((aligned(32))) _Float16 sB[4 * 32 * 16];  // 4 KB

    const int tid  = threadIdx.x;
    const int lane = tid & 31;
    const int wv   = tid >> 5;                 // 0..7 (wave32)
    const int wvM  = wv >> 1;                  // 0..3 : m0 = wvM*16
    const int wvN  = wv & 1;                   // 0..1 : n0 = wvN*32
    const int gm0  = blockIdx.x * TILE_M;
    const int gn0  = blockIdx.y * TILE_N;

    // ---- A staging: 64 rows x 32 k, 8 contiguous k per thread ----
    const int arow   = tid >> 2;               // 0..63
    const int acol0  = (tid & 3) * 8;          // 0,8,16,24 (fixed hi/group)
    const int aWave  = arow >> 4;              // which wave-row tile
    const int am     = arow & 15;
    const int aLane  = frag_lane(am, acol0);   // constant over the 8-block
    const int aIdx0  = frag_idx(acol0);        // 0 or 8
    const int gm     = gm0 + arow;             // global row
    const int angle  = gm >> 8;
    const int vdet   = gm & 255;
    const float vv   = (float)vdet - 0.5f * (float)(DET_V - 1);

    // ---- B staging: 32 k x 64 cols, 8 contiguous cols per thread ----
    const int brow   = tid >> 3;               // k within tile: 0..31
    const int bcol0  = (tid & 7) * 8;          // 0..56
    const int bIdx   = frag_idx(brow);         // fixed per thread
    const int bHi    = 16 * ((brow >> 3) & 1); // lane offset from k

    v8f acc0 = {}, acc1 = {};

    for (int k0 = 0; k0 < DET_U; k0 += TILE_K) {
        // Prefetch next A tile's row segment (gfx1250 global_prefetch path).
        if (k0 + TILE_K < DET_U)
            __builtin_prefetch(&proj[(size_t)gm * DET_U + k0 + TILE_K], 0, 0);

        // ---- stage A (fused cosine + redundancy pre-weight), one 16B store ----
        v8h aval;
#pragma unroll
        for (int j = 0; j < 8; ++j) {
            const int k    = k0 + acol0 + j;
            const float uu = (float)k - 0.5f * (float)(DET_U - 1);
            const float cw = D_SD_F * rsqrtf(D_SD_F * D_SD_F + uu * uu + vv * vv);
            aval[j] = (_Float16)(proj[(size_t)gm * DET_U + k] * cw
                                 * redw[angle * DET_U + k]);
        }
        *(v8h*)&sA[(aWave * 32 + aLane) * 16 + aIdx0] = aval;

        // ---- stage B into fragment order (scatter across lanes) ----
#pragma unroll
        for (int j = 0; j < 8; ++j) {
            const int col   = bcol0 + j;
            const int bWave = col >> 4;
            const int bn    = col & 15;
            sB[(bWave * 32 + bn + bHi) * 16 + bIdx] =
                H[(size_t)(k0 + brow) * DET_U + gn0 + col];
        }
        __syncthreads();

        // ---- fragments: one contiguous 32B read each (2x ds_load_b128) ----
        const v16h a  = *(const v16h*)&sA[(wvM * 32 + lane) * 16];
        const v16h b0 = *(const v16h*)&sB[((wvN * 2 + 0) * 32 + lane) * 16];
        const v16h b1 = *(const v16h*)&sB[((wvN * 2 + 1) * 32 + lane) * 16];

        acc0 = __builtin_amdgcn_wmma_f32_16x16x32_f16(
                   false, a, false, b0, (short)0, acc0, false, false);
        acc1 = __builtin_amdgcn_wmma_f32_16x16x32_f16(
                   false, a, false, b1, (short)0, acc1, false, false);
        __syncthreads();
    }

    // ---- store C (VGPR r: M = r + 8*(lane>>4), N = lane&15) ----
    const int col = lane & 15;
    const int rhi = (lane >> 4) * 8;
    const size_t rowBase = (size_t)(gm0 + wvM * 16 + rhi);
#pragma unroll
    for (int r = 0; r < 8; ++r) {
        float* o = out + (rowBase + r) * DET_U + gn0 + wvN * 32 + col;
        o[0]  = acc0[r];
        o[16] = acc1[r];
    }
}

// ---------------------------------------------------------------------------
// Kernel 4: backprojection.  Block = 128 threads along x (coalesced writes),
// (y, z) = (blockIdx.x, blockIdx.y).  Shared sin/cos table for 256 angles.
// ---------------------------------------------------------------------------
__global__ __launch_bounds__(128)
void backproject(const float* __restrict__ filt, float* __restrict__ vol)
{
    __shared__ float2 trig[NANG];
    for (int i = threadIdx.x; i < NANG; i += blockDim.x) {
        float s, c;
        __sincosf((float)i * (2.0f * PI_F / (float)NANG), &s, &c);
        trig[i] = make_float2(c, s);
    }
    __syncthreads();

    const int x = threadIdx.x;
    const int y = blockIdx.x;
    const int z = blockIdx.y;
    const float xc = (float)x - 0.5f * (float)(VOLN - 1);
    const float yc = (float)y - 0.5f * (float)(VOLN - 1);
    const float zc = (float)z - 0.5f * (float)(VOLN - 1);
    const float cu = 0.5f * (float)(DET_U - 1);
    const float cv = 0.5f * (float)(DET_V - 1);

    float acc = 0.0f;
    for (int a = 0; a < NANG; ++a) {
        const float c = trig[a].x, s = trig[a].y;
        const float t    = -xc * s + yc * c;
        const float r    = D_SO_F - (xc * c + yc * s);
        const float rinv = 1.0f / r;
        const float iu   = D_SD_F * t * rinv + cu;
        const float iv   = D_SD_F * zc * rinv + cv;
        float wr = D_SO_F * rinv;  wr *= wr;

        const float u0f = floorf(iu);
        const float v0f = floorf(iv);
        const bool valid = (u0f >= 0.0f) && (u0f <= (float)(DET_U - 2)) &&
                           (v0f >= 0.0f) && (v0f <= (float)(DET_V - 2));
        const int u0 = (int)fminf(fmaxf(u0f, 0.0f), (float)(DET_U - 2));
        const int v0 = (int)fminf(fmaxf(v0f, 0.0f), (float)(DET_V - 2));
        const float fu = iu - u0f;
        const float fv = iv - v0f;

        const float* p = filt + ((size_t)a * DET_V + v0) * DET_U + u0;
        const float p00 = p[0];
        const float p01 = p[1];
        const float p10 = p[DET_U];
        const float p11 = p[DET_U + 1];
        const float interp = p00 * (1.0f - fv) * (1.0f - fu)
                           + p01 * (1.0f - fv) * fu
                           + p10 * fv * (1.0f - fu)
                           + p11 * fv * fu;
        acc += valid ? interp * wr : 0.0f;
    }
    vol[((size_t)z * VOLN + y) * VOLN + x] = acc * (PI_F / (float)NANG);
}

// ---------------------------------------------------------------------------
extern "C" void kernel_launch(void* const* d_in, const int* in_sizes, int n_in,
                              void* d_out, int out_size, void* d_ws, size_t ws_size,
                              hipStream_t stream) {
    const float* proj = (const float*)d_in[0];   // (1, 256, 256, 384) f32
    const float* ramp = (const float*)d_in[1];   // (193,) f32
    const float* redw = (const float*)d_in[2];   // (256, 384) f32
    float* vol = (float*)d_out;                  // (1, 128, 128, 128) f32

    char* ws = (char*)d_ws;
    float*     h    = (float*)(ws + OFF_H);
    _Float16*  Hmat = (_Float16*)(ws + OFF_HM);
    float*     filt = (float*)(ws + OFF_FILT);

    (void)in_sizes; (void)n_in; (void)out_size; (void)ws_size;

    build_h   <<<1,   DET_U, 0, stream>>>(ramp, h);
    build_Hmat<<<DET_U, DET_U, 0, stream>>>(h, Hmat);

    dim3 gGemm((NANG * DET_V) / TILE_M, DET_U / TILE_N);   // (1024, 6)
    filter_gemm<<<gGemm, 256, 0, stream>>>(proj, redw, Hmat, filt);

    dim3 gBp(VOLN, VOLN);                                  // (128, 128)
    backproject<<<gBp, VOLN, 0, stream>>>(filt, vol);
}
